// EdgePredictionNetwork_58815282151679
// MI455X (gfx1250) — matrix operand order repaired and checked by reading, not compiled, not run on hardware
//
#include <hip/hip_runtime.h>
#include <hip/hip_bf16.h>
#include <cstdint>
#include <cstddef>

typedef __bf16 bf16;
typedef __attribute__((ext_vector_type(16))) __bf16 bf16x16;
typedef __attribute__((ext_vector_type(8)))  float  f32x8;

// ---------------------------------------------------------------- constants
#define kNN    2048            // nodes
#define kNB    64              // molecules
#define kMOL   32              // nodes / molecule
#define kEPM   992             // edges / molecule = 32*31
#define kEE    63488           // total edges
#define kSD    256             // SDIM
#define kVD    64              // VDIM
#define kED    32              // EDIM
#define kFD    16              // FDIM
#define kL     5               // layers
#define kDIN   546             // 2*256+32+2
#define kK1    576             // padded din
#define kNCAT  352             // 256+64+32
#define kKB0   288             // padded 257
#define kNB1   32              // padded 5
#define kLDP   34              // LDS row pitch (32 + 2 pad, bank-conflict break)

// fast silu: exp + rcp (v_rcp_f32) instead of the IEEE division chain
__device__ __forceinline__ float silu_f(float x) {
  return x * __builtin_amdgcn_rcpf(1.0f + __expf(-x));
}

// A fragment (16x32 bf16, ISA 7.12.2): lanes 0-15 -> K{0..7,16..23}, lanes 16-31 -> K{8..15,24..31}
__device__ __forceinline__ bf16x16 frag_a(const bf16* rowbase, int lane) {
  const uint32_t* p = (const uint32_t*)rowbase;
  const int kb2 = (lane < 16) ? 0 : 4;         // K-pair base (K=0 or K=8)
  union { uint32_t u[8]; bf16x16 v; } c;
#pragma unroll
  for (int i = 0; i < 4; ++i) c.u[i] = p[kb2 + i];          // K = kb .. kb+7
#pragma unroll
  for (int i = 0; i < 4; ++i) c.u[4 + i] = p[kb2 + 8 + i];  // K = kb+16 .. kb+23
  return c.v;
}

// B fragment (32x16 bf16): lanes 0-15 -> K{0..7,16..23}, lanes 16-31 -> K{8..15,24..31}
// (LDS tile stored N-major: Blds[n*kLDP + k], so K runs contiguous per lane)
__device__ __forceinline__ bf16x16 frag_b(const bf16* rowbase, int lane) {
  const uint32_t* p = (const uint32_t*)rowbase;
  const int kb2 = (lane < 16) ? 0 : 4;         // K-pair base (K=0 or K=8)
  union { uint32_t u[8]; bf16x16 v; } c;
#pragma unroll
  for (int i = 0; i < 4; ++i) c.u[i] = p[kb2 + i];
#pragma unroll
  for (int i = 0; i < 4; ++i) c.u[4 + i] = p[kb2 + 8 + i];
  return c.v;
}

// ---------------------------------------------------------------- generic WMMA GEMM
// C[M,N] = act(A[M,K] @ B[K,N] + bias) ; A,B bf16 row-major.
// Block tile 128x32, 8 waves; each wave owns two M-adjacent 16x16 tiles sharing one B frag
// -> 2 v_wmma per wave per k-step.
__global__ __launch_bounds__(256)
void gemm_bf16_kernel(const bf16* __restrict__ A, const bf16* __restrict__ Bm,
                      const float* __restrict__ bias, float* __restrict__ Cf,
                      bf16* __restrict__ Cb, int M, int N, int K, int act) {
  __shared__ bf16 Alds[128 * kLDP];
  __shared__ bf16 Blds[32 * kLDP];
  const int tid  = threadIdx.x;
  const int lane = tid & 31;
  const int w    = tid >> 5;
  const int nt   = w & 1;            // 0..1  (N tile)
  const int mtb  = (w >> 1) * 2;     // 0,2,4,6 (M tile pair base)
  const int r0   = blockIdx.x * 128;
  const int n0   = blockIdx.y * 32;
  const int rl   = tid >> 1;         // 0..127 (A staging row)
  const int kc   = (tid & 1) * 16;   // A staging K half-row
  const int bn   = tid & 31;         // B staging column
  const int bk0  = tid >> 5;         // B staging K base

  f32x8 acc0 = {0.f, 0.f, 0.f, 0.f, 0.f, 0.f, 0.f, 0.f};
  f32x8 acc1 = {0.f, 0.f, 0.f, 0.f, 0.f, 0.f, 0.f, 0.f};

  for (int k0 = 0; k0 < K; k0 += 32) {
    // stage A tile (128x32) -> row-major padded
    const uint32_t* as = (const uint32_t*)(A + (size_t)(r0 + rl) * K + k0 + kc);
    uint32_t* ad = (uint32_t*)(Alds + rl * kLDP + kc);
#pragma unroll
    for (int i = 0; i < 8; ++i) ad[i] = as[i];
    if (k0 + 32 < K) __builtin_prefetch(as + (size_t)16, 0, 1);  // next K tile of this row
    // stage B tile (32x32) -> N-major padded
#pragma unroll
    for (int it = 0; it < 4; ++it) {
      int kl = bk0 + it * 8;
      Blds[bn * kLDP + kl] = Bm[(size_t)(k0 + kl) * N + n0 + bn];
    }
    __syncthreads();
    bf16x16 af0 = frag_a(Alds + (mtb * 16 + (lane & 15)) * kLDP, lane);
    bf16x16 af1 = frag_a(Alds + ((mtb + 1) * 16 + (lane & 15)) * kLDP, lane);
    bf16x16 bfr = frag_b(Blds + (nt * 16 + (lane & 15)) * kLDP, lane);
    acc0 = __builtin_amdgcn_wmma_f32_16x16x32_bf16(false, af0, false, bfr,
                                                   (short)0, acc0, false, false);
    acc1 = __builtin_amdgcn_wmma_f32_16x16x32_bf16(false, af1, false, bfr,
                                                   (short)0, acc1, false, false);
    __syncthreads();
  }

  const int ng = n0 + nt * 16 + (lane & 15);
  const float bv = bias ? bias[ng] : 0.0f;
  const int mb0 = r0 + mtb * 16 + ((lane >= 16) ? 8 : 0);
#pragma unroll
  for (int i = 0; i < 8; ++i) {
    float xv = acc0[i] + bv;
    if (act) xv = silu_f(xv);
    size_t o = (size_t)(mb0 + i) * N + ng;
    if (Cf) Cf[o] = xv;
    if (Cb) Cb[o] = (bf16)xv;
  }
#pragma unroll
  for (int i = 0; i < 8; ++i) {
    float xv = acc1[i] + bv;
    if (act) xv = silu_f(xv);
    size_t o = (size_t)(mb0 + 16 + i) * N + ng;
    if (Cf) Cf[o] = xv;
    if (Cb) Cb[o] = (bf16)xv;
  }
}

// ---------------------------------------------------------------- fused edge-feature WMMA GEMM
// h = silu([s[src]|s[dst]|e|d|a|0pad] @ W1 + b1) ; M=kEE, N=kSD, K=kK1. Output bf16.
__global__ __launch_bounds__(256)
void gemm_edge_feat_kernel(const bf16* __restrict__ s_b, const bf16* __restrict__ e_b,
                           const float* __restrict__ dbuf, const float* __restrict__ abuf,
                           const int* __restrict__ srcI, const int* __restrict__ dstI,
                           const bf16* __restrict__ W, const float* __restrict__ bias,
                           bf16* __restrict__ h_b) {
  __shared__ bf16 Alds[128 * kLDP];
  __shared__ bf16 Blds[32 * kLDP];
  const int tid  = threadIdx.x;
  const int lane = tid & 31;
  const int w    = tid >> 5;
  const int nt   = w & 1;
  const int mtb  = (w >> 1) * 2;
  const int r0   = blockIdx.x * 128;
  const int n0   = blockIdx.y * 32;
  const int rl   = tid >> 1;
  const int kc   = (tid & 1) * 16;
  const int bn   = tid & 31;
  const int bk0  = tid >> 5;

  const int r  = r0 + rl;
  const int sN = srcI[r];
  const int dN = dstI[r];

  f32x8 acc0 = {0.f, 0.f, 0.f, 0.f, 0.f, 0.f, 0.f, 0.f};
  f32x8 acc1 = {0.f, 0.f, 0.f, 0.f, 0.f, 0.f, 0.f, 0.f};

  for (int k0 = 0; k0 < kK1; k0 += 32) {
    // gather-stage A tile row r, columns k0+kc .. k0+kc+15
#pragma unroll
    for (int i = 0; i < 16; ++i) {
      const int k = k0 + kc + i;
      bf16 v;
      if (k < kSD)                v = s_b[(size_t)sN * kSD + k];
      else if (k < 2 * kSD)       v = s_b[(size_t)dN * kSD + (k - kSD)];
      else if (k < 2 * kSD + kED) v = e_b[(size_t)r * kED + (k - 2 * kSD)];
      else if (k == kDIN - 2)     v = (bf16)dbuf[r];
      else if (k == kDIN - 1)     v = (bf16)abuf[r];
      else                        v = (bf16)0.0f;
      Alds[rl * kLDP + kc + i] = v;
    }
#pragma unroll
    for (int it = 0; it < 4; ++it) {
      int kl = bk0 + it * 8;
      Blds[bn * kLDP + kl] = W[(size_t)(k0 + kl) * kSD + n0 + bn];
    }
    __syncthreads();
    bf16x16 af0 = frag_a(Alds + (mtb * 16 + (lane & 15)) * kLDP, lane);
    bf16x16 af1 = frag_a(Alds + ((mtb + 1) * 16 + (lane & 15)) * kLDP, lane);
    bf16x16 bfr = frag_b(Blds + (nt * 16 + (lane & 15)) * kLDP, lane);
    acc0 = __builtin_amdgcn_wmma_f32_16x16x32_bf16(false, af0, false, bfr,
                                                   (short)0, acc0, false, false);
    acc1 = __builtin_amdgcn_wmma_f32_16x16x32_bf16(false, af1, false, bfr,
                                                   (short)0, acc1, false, false);
    __syncthreads();
  }

  const int ng = n0 + nt * 16 + (lane & 15);
  const float bv = bias[ng];
  const int mb0 = r0 + mtb * 16 + ((lane >= 16) ? 8 : 0);
#pragma unroll
  for (int i = 0; i < 8; ++i)
    h_b[(size_t)(mb0 + i) * kSD + ng] = (bf16)silu_f(acc0[i] + bv);
#pragma unroll
  for (int i = 0; i < 8; ++i)
    h_b[(size_t)(mb0 + 16 + i) * kSD + ng] = (bf16)silu_f(acc1[i] + bv);
}

// ---------------------------------------------------------------- small kernels
__global__ void center_kernel(const float* __restrict__ in, float* __restrict__ out) {
  const int b = blockIdx.x, i = threadIdx.x;          // 64 blocks x 32 lanes
  const int node = b * kMOL + i;
  float x = in[node * 3 + 0], y = in[node * 3 + 1], z = in[node * 3 + 2];
  float sx = x, sy = y, sz = z;
#pragma unroll
  for (int m = 16; m >= 1; m >>= 1) {
    sx += __shfl_xor(sx, m, 32);
    sy += __shfl_xor(sy, m, 32);
    sz += __shfl_xor(sz, m, 32);
  }
  const float inv = 1.0f / (float)kMOL;
  out[node * 3 + 0] = x - sx * inv;
  out[node * 3 + 1] = y - sy * inv;
  out[node * 3 + 2] = z - sz * inv;
}

__global__ void embed_node_kernel(const float* __restrict__ x, const float* __restrict__ t,
                                  const int* __restrict__ batch,
                                  const float* __restrict__ W_tma, const float* __restrict__ b_tma,
                                  const float* __restrict__ W_am, const float* __restrict__ b_am,
                                  bf16* __restrict__ out) {
  const int tid = blockIdx.x * blockDim.x + threadIdx.x;
  if (tid >= kNN * kSD) return;
  const int node = tid / kSD, c = tid % kSD;
  float acc = b_am[c] + t[batch[node]] * W_tma[c] + b_tma[c];
  const float* xr = x + (size_t)node * kFD;
#pragma unroll
  for (int f = 0; f < kFD; ++f) acc += xr[f] * W_am[f * kSD + c];
  out[tid] = (bf16)acc;
}

__global__ void embed_edge_kernel(const float* __restrict__ x, const float* __restrict__ t,
                                  const int* __restrict__ bedge, const int* __restrict__ dstI,
                                  const float* __restrict__ W_tmb, const float* __restrict__ b_tmb,
                                  const float* __restrict__ W_bm, const float* __restrict__ b_bm,
                                  bf16* __restrict__ out) {
  const int tid = blockIdx.x * blockDim.x + threadIdx.x;
  if (tid >= kEE * kED) return;
  const int r = tid / kED, c = tid % kED;
  float acc = b_bm[c] + t[bedge[r]] * W_tmb[c] + b_tmb[c];
  const float* xr = x + (size_t)dstI[r] * kFD;
#pragma unroll
  for (int f = 0; f < kFD; ++f) acc += xr[f] * W_bm[f * kED + c];
  out[tid] = (bf16)acc;
}

__global__ void edge_geom_kernel(const float* __restrict__ p, const int* __restrict__ srcI,
                                 const int* __restrict__ dstI, float* __restrict__ dbuf,
                                 float* __restrict__ abuf, float* __restrict__ rn) {
  const int r = blockIdx.x * blockDim.x + threadIdx.x;
  if (r >= kEE) return;
  const float* ps = p + (size_t)srcI[r] * 3;
  const float* pd = p + (size_t)dstI[r] * 3;
  const float rx = pd[0] - ps[0], ry = pd[1] - ps[1], rz = pd[2] - ps[2];
  const float a = pd[0] * ps[0] + pd[1] * ps[1] + pd[2] * ps[2];
  const float d = sqrtf(fmaxf(rx * rx + ry * ry + rz * rz, 1e-6f));
  const float inv = __builtin_amdgcn_rcpf(1.0f + d);
  dbuf[r] = d; abuf[r] = a;
  rn[r * 3 + 0] = rx * inv; rn[r * 3 + 1] = ry * inv; rn[r * 3 + 2] = rz * inv;
}

__global__ void gp_kernel(const bf16* __restrict__ h, const float* __restrict__ wp,
                          float* __restrict__ gp) {
  const int r = blockIdx.x * blockDim.x + threadIdx.x;
  if (r >= kEE) return;
  float acc = 0.f;
  const bf16* hr = h + (size_t)r * kSD;
  for (int c = 0; c < kSD; ++c) acc += (float)hr[c] * wp[c];
  gp[r] = tanhf(acc);
}

// s += mean_{incoming}(z[:, 0:256]); refresh s_bf16. Fully-connected topology:
// edges of molecule b contiguous; dst i receives from j!=i at j*31 + i - (i>j).
__global__ void agg_s_kernel(const bf16* __restrict__ z, float* __restrict__ s,
                             bf16* __restrict__ s_b) {
  const int tid = blockIdx.x * blockDim.x + threadIdx.x;
  if (tid >= kNN * kSD) return;
  const int node = tid / kSD, c = tid % kSD;
  const int b = node / kMOL, i = node % kMOL;
  const size_t base = (size_t)b * kEPM;
  float acc = 0.f;
  for (int j = 0; j < kMOL; ++j) {
    if (j == i) continue;
    const int idx = j * (kMOL - 1) + i - ((i > j) ? 1 : 0);
    acc += (float)z[(base + idx) * kNCAT + c];
  }
  const float v = s[tid] + acc * (1.0f / (kMOL - 1));
  s[tid] = v; s_b[tid] = (bf16)v;
}

__global__ void agg_v_kernel(const bf16* __restrict__ z, const float* __restrict__ rn,
                             float* __restrict__ v) {
  const int tid = blockIdx.x * blockDim.x + threadIdx.x;
  if (tid >= kNN * kVD) return;
  const int node = tid / kVD, c = tid % kVD;
  const int b = node / kMOL, i = node % kMOL;
  const size_t base = (size_t)b * kEPM;
  float a0 = 0.f, a1 = 0.f, a2 = 0.f;
  for (int j = 0; j < kMOL; ++j) {
    if (j == i) continue;
    const size_t e = base + j * (kMOL - 1) + i - ((i > j) ? 1 : 0);
    const float m = (float)z[e * kNCAT + kSD + c];
    a0 += rn[e * 3 + 0] * m; a1 += rn[e * 3 + 1] * m; a2 += rn[e * 3 + 2] * m;
  }
  const float sc = 1.0f / (kMOL - 1);
  v[((size_t)node * 3 + 0) * kVD + c] += a0 * sc;
  v[((size_t)node * 3 + 1) * kVD + c] += a1 * sc;
  v[((size_t)node * 3 + 2) * kVD + c] += a2 * sc;
}

__global__ void e_upd_kernel(const bf16* __restrict__ z, float* __restrict__ e,
                             bf16* __restrict__ e_b) {
  const int tid = blockIdx.x * blockDim.x + threadIdx.x;
  if (tid >= kEE * kED) return;
  const int r = tid / kED, c = tid % kED;
  const float v = e[tid] + (float)z[(size_t)r * kNCAT + kSD + kVD + c];
  e[tid] = v; e_b[tid] = (bf16)v;
}

__global__ void agg_p_kernel(const float* __restrict__ rn, const float* __restrict__ gp,
                             float* __restrict__ p) {
  const int tid = blockIdx.x * blockDim.x + threadIdx.x;
  if (tid >= kNN * 3) return;
  const int node = tid / 3, k = tid % 3;
  const int b = node / kMOL, i = node % kMOL;
  const size_t base = (size_t)b * kEPM;
  float acc = 0.f;
  for (int j = 0; j < kMOL; ++j) {
    if (j == i) continue;
    const size_t e = base + j * (kMOL - 1) + i - ((i > j) ? 1 : 0);
    acc += rn[e * 3 + k] * gp[e];
  }
  p[tid] += acc * (1.0f / (kMOL - 1));
}

__global__ void coords_kernel(const float* __restrict__ p, const float* __restrict__ v,
                              const float* __restrict__ W_cl, float* __restrict__ coords) {
  const int tid = blockIdx.x * blockDim.x + threadIdx.x;
  if (tid >= kNN * 3) return;
  const int node = tid / 3, k = tid % 3;
  float acc = p[tid];
  const float* vr = v + ((size_t)node * 3 + k) * kVD;
  for (int c = 0; c < kVD; ++c) acc += vr[c] * W_cl[c];
  coords[tid] = acc;
}

__global__ void esym_kernel(const float* __restrict__ e, bf16* __restrict__ esym) {
  const int tid = blockIdx.x * blockDim.x + threadIdx.x;
  if (tid >= kEE * kED) return;
  const int r = tid / kED, c = tid % kED;
  const int b = r / kEPM, loc = r % kEPM;
  const int j = loc / (kMOL - 1), rem = loc % (kMOL - 1);
  const int i = rem + ((rem >= j) ? 1 : 0);
  const int rr = b * kEPM + i * (kMOL - 1) + j - ((j > i) ? 1 : 0);
  esym[tid] = (bf16)(0.5f * (e[tid] + e[(size_t)rr * kED + c]));
}

__global__ void combine_kernel(const bf16* __restrict__ f0, const float* __restrict__ sh,
                               const int* __restrict__ srcI, const int* __restrict__ dstI,
                               const float* __restrict__ coords, bf16* __restrict__ edge_in) {
  const int tid = blockIdx.x * blockDim.x + threadIdx.x;
  if (tid >= kEE * kKB0) return;
  const int r = tid / kKB0, c = tid % kKB0;
  float val;
  if (c < kSD) {
    val = (float)f0[(size_t)r * kSD + c] + sh[(size_t)dstI[r] * kSD + c] +
          sh[(size_t)srcI[r] * kSD + c];
  } else if (c == kSD) {
    const float* cs = coords + (size_t)srcI[r] * 3;
    const float* cd = coords + (size_t)dstI[r] * 3;
    const float dx = cd[0] - cs[0], dy = cd[1] - cs[1], dz = cd[2] - cs[2];
    val = dx * dx + dy * dy + dz * dz;
  } else {
    val = 0.f;
  }
  edge_in[tid] = (bf16)val;
}

__global__ void out_copy_kernel(const float* __restrict__ o32, float* __restrict__ out) {
  const int tid = blockIdx.x * blockDim.x + threadIdx.x;
  if (tid >= kEE * 5) return;
  const int r = tid / 5, c = tid % 5;
  out[tid] = o32[(size_t)r * kNB1 + c];
}

// ---------------------------------------------------------------- weight packing
__global__ void cvt_bf16_kernel(const float* __restrict__ in, bf16* __restrict__ out, int n) {
  const int tid = blockIdx.x * blockDim.x + threadIdx.x;
  if (tid < n) out[tid] = (bf16)in[tid];
}

__global__ void pack_w1_kernel(const float* __restrict__ W1, bf16* __restrict__ out) {
  const int tid = blockIdx.x * blockDim.x + threadIdx.x;
  if (tid >= kL * kK1 * kSD) return;
  const int l = tid / (kK1 * kSD), rem = tid % (kK1 * kSD);
  const int k = rem / kSD, c = rem % kSD;
  out[tid] = (k < kDIN) ? (bf16)W1[((size_t)l * kDIN + k) * kSD + c] : (bf16)0.0f;
}

__global__ void pack_wcat_kernel(const float* __restrict__ Ws, const float* __restrict__ Wv,
                                 const float* __restrict__ We, bf16* __restrict__ out) {
  const int tid = blockIdx.x * blockDim.x + threadIdx.x;
  if (tid >= kL * kSD * kNCAT) return;
  const int l = tid / (kSD * kNCAT), rem = tid % (kSD * kNCAT);
  const int k = rem / kNCAT, c = rem % kNCAT;
  float v;
  if (c < kSD)            v = Ws[((size_t)l * kSD + k) * kSD + c];
  else if (c < kSD + kVD) v = Wv[((size_t)l * kSD + k) * kVD + (c - kSD)];
  else                    v = We[((size_t)l * kSD + k) * kED + (c - kSD - kVD)];
  out[tid] = (bf16)v;
}

__global__ void pack_wb0_kernel(const float* __restrict__ W, bf16* __restrict__ out) {
  const int tid = blockIdx.x * blockDim.x + threadIdx.x;
  if (tid >= kKB0 * kSD) return;
  const int k = tid / kSD, c = tid % kSD;
  out[tid] = (k < kSD + 1) ? (bf16)W[(size_t)k * kSD + c] : (bf16)0.0f;
}

__global__ void pack_wb1_kernel(const float* __restrict__ W, const float* __restrict__ bias,
                                bf16* __restrict__ outW, float* __restrict__ outB) {
  const int tid = blockIdx.x * blockDim.x + threadIdx.x;
  if (tid >= kSD * kNB1) return;
  const int k = tid / kNB1, c = tid % kNB1;
  outW[tid] = (c < 5) ? (bf16)W[(size_t)k * 5 + c] : (bf16)0.0f;
  if (tid < kNB1) outB[tid] = (tid < 5) ? bias[tid] : 0.0f;
}

// ---------------------------------------------------------------- host
static inline dim3 grid1(long long n) { return dim3((unsigned)((n + 255) / 256)); }

extern "C" void kernel_launch(void* const* d_in, const int* in_sizes, int n_in,
                              void* d_out, int out_size, void* d_ws, size_t ws_size,
                              hipStream_t stream) {
  (void)in_sizes; (void)n_in; (void)out_size; (void)ws_size;
  const float* x      = (const float*)d_in[0];
  const float* t      = (const float*)d_in[1];
  const float* pos    = (const float*)d_in[2];
  const int*   eig    = (const int*)d_in[3];
  const int*   batch  = (const int*)d_in[4];
  const int*   bedge  = (const int*)d_in[5];
  const float* W_tma  = (const float*)d_in[6];
  const float* b_tma  = (const float*)d_in[7];
  const float* W_tmb  = (const float*)d_in[8];
  const float* b_tmb  = (const float*)d_in[9];
  const float* W_am   = (const float*)d_in[10];
  const float* b_am   = (const float*)d_in[11];
  const float* W_bm   = (const float*)d_in[12];
  const float* b_bm   = (const float*)d_in[13];
  const float* W_atm  = (const float*)d_in[14];
  const float* b_atm  = (const float*)d_in[15];
  const float* W_btm  = (const float*)d_in[16];
  const float* b_btm  = (const float*)d_in[17];
  const float* gW1    = (const float*)d_in[18];
  const float* gb1    = (const float*)d_in[19];
  const float* gWs    = (const float*)d_in[20];
  const float* gWv    = (const float*)d_in[21];
  const float* gWe    = (const float*)d_in[22];
  const float* gwp    = (const float*)d_in[23];
  const float* W_sh   = (const float*)d_in[24];
  const float* b_sh   = (const float*)d_in[25];
  const float* W_bmap = (const float*)d_in[26];
  const float* b_bmap = (const float*)d_in[27];
  const float* W_b0   = (const float*)d_in[28];
  const float* b_b0   = (const float*)d_in[29];
  const float* W_b1   = (const float*)d_in[30];
  const float* b_b1   = (const float*)d_in[31];
  const float* W_cl   = (const float*)d_in[32];
  const int* srcI = eig;
  const int* dstI = eig + kEE;
  float* out = (float*)d_out;

  // ---- workspace layout
  char* wsb = (char*)d_ws;
  size_t off = 0;
  auto alloc = [&](size_t bytes) -> void* {
    size_t o = (off + 255) & ~(size_t)255;
    off = o + bytes;
    return (void*)(wsb + o);
  };
  bf16*  w1p    = (bf16*)alloc((size_t)kL * kK1 * kSD * 2);
  bf16*  wcat   = (bf16*)alloc((size_t)kL * kSD * kNCAT * 2);
  bf16*  watm   = (bf16*)alloc((size_t)kSD * kSD * 2);
  bf16*  wbtm   = (bf16*)alloc((size_t)kED * kED * 2);
  bf16*  wshb   = (bf16*)alloc((size_t)kSD * kSD * 2);
  bf16*  wbmapb = (bf16*)alloc((size_t)kED * kSD * 2);
  bf16*  wb0b   = (bf16*)alloc((size_t)kKB0 * kSD * 2);
  bf16*  wb1b   = (bf16*)alloc((size_t)kSD * kNB1 * 2);
  float* bb1p   = (float*)alloc((size_t)kNB1 * 4);
  float* s_f    = (float*)alloc((size_t)kNN * kSD * 4);
  bf16*  s_b    = (bf16*)alloc((size_t)kNN * kSD * 2);
  float* e_f    = (float*)alloc((size_t)kEE * kED * 4);
  bf16*  e_b    = (bf16*)alloc((size_t)kEE * kED * 2);
  float* p_f    = (float*)alloc((size_t)kNN * 3 * 4);
  float* v_f    = (float*)alloc((size_t)kNN * 3 * kVD * 4);
  float* dbuf   = (float*)alloc((size_t)kEE * 4);
  float* abuf   = (float*)alloc((size_t)kEE * 4);
  float* rn     = (float*)alloc((size_t)kEE * 3 * 4);
  bf16*  h_b    = (bf16*)alloc((size_t)kEE * kSD * 2);
  bf16*  z_b    = (bf16*)alloc((size_t)kEE * kNCAT * 2);
  float* gp     = (float*)alloc((size_t)kEE * 4);
  bf16*  tmpn_b = (bf16*)alloc((size_t)kNN * kSD * 2);
  bf16*  tmpe_b = (bf16*)alloc((size_t)kEE * kED * 2);
  float* sh_f   = (float*)alloc((size_t)kNN * kSD * 4);
  float* coords = (float*)alloc((size_t)kNN * 3 * 4);
  bf16*  esym_b = (bf16*)alloc((size_t)kEE * kED * 2);
  bf16*  ein_b  = (bf16*)alloc((size_t)kEE * kKB0 * 2);
  bf16*  f0_b   = h_b;                       // reuse: h dead at head
  bf16*  g_b    = h_b;                       // reuse: f0 dead after combine
  float* out32  = e_f;                       // reuse: e dead after esym

  // ---- pack / convert weights
  pack_w1_kernel<<<grid1((long long)kL * kK1 * kSD), 256, 0, stream>>>(gW1, w1p);
  pack_wcat_kernel<<<grid1((long long)kL * kSD * kNCAT), 256, 0, stream>>>(gWs, gWv, gWe, wcat);
  cvt_bf16_kernel<<<grid1(kSD * kSD), 256, 0, stream>>>(W_atm, watm, kSD * kSD);
  cvt_bf16_kernel<<<grid1(kED * kED), 256, 0, stream>>>(W_btm, wbtm, kED * kED);
  cvt_bf16_kernel<<<grid1(kSD * kSD), 256, 0, stream>>>(W_sh, wshb, kSD * kSD);
  cvt_bf16_kernel<<<grid1(kED * kSD), 256, 0, stream>>>(W_bmap, wbmapb, kED * kSD);
  pack_wb0_kernel<<<grid1(kKB0 * kSD), 256, 0, stream>>>(W_b0, wb0b);
  pack_wb1_kernel<<<grid1(kSD * kNB1), 256, 0, stream>>>(W_b1, b_b1, wb1b, bb1p);

  // ---- init: center positions, zero v
  center_kernel<<<kNB, kMOL, 0, stream>>>(pos, p_f);
  hipMemsetAsync(v_f, 0, (size_t)kNN * 3 * kVD * 4, stream);

  // ---- embeddings: s = (x@W_am+b_am+tnode)@W_atm+b_atm ; e analog
  embed_node_kernel<<<grid1(kNN * kSD), 256, 0, stream>>>(x, t, batch, W_tma, b_tma,
                                                          W_am, b_am, tmpn_b);
  gemm_bf16_kernel<<<dim3(kNN / 128, kSD / 32), 256, 0, stream>>>(
      tmpn_b, watm, b_atm, s_f, s_b, kNN, kSD, kSD, 0);
  embed_edge_kernel<<<grid1((long long)kEE * kED), 256, 0, stream>>>(
      x, t, bedge, dstI, W_tmb, b_tmb, W_bm, b_bm, tmpe_b);
  gemm_bf16_kernel<<<dim3(kEE / 128, kED / 32), 256, 0, stream>>>(
      tmpe_b, wbtm, b_btm, e_f, e_b, kEE, kED, kED, 0);

  // ---- message passing layers
  for (int l = 0; l < kL; ++l) {
    edge_geom_kernel<<<grid1(kEE), 256, 0, stream>>>(p_f, srcI, dstI, dbuf, abuf, rn);
    gemm_edge_feat_kernel<<<dim3(kEE / 128, kSD / 32), 256, 0, stream>>>(
        s_b, e_b, dbuf, abuf, srcI, dstI, w1p + (size_t)l * kK1 * kSD,
        gb1 + (size_t)l * kSD, h_b);
    gemm_bf16_kernel<<<dim3(kEE / 128, kNCAT / 32), 256, 0, stream>>>(
        h_b, wcat + (size_t)l * kSD * kNCAT, nullptr, nullptr, z_b, kEE, kNCAT, kSD, 0);
    gp_kernel<<<grid1(kEE), 256, 0, stream>>>(h_b, gwp + (size_t)l * kSD, gp);
    agg_s_kernel<<<grid1(kNN * kSD), 256, 0, stream>>>(z_b, s_f, s_b);
    agg_v_kernel<<<grid1(kNN * kVD), 256, 0, stream>>>(z_b, rn, v_f);
    e_upd_kernel<<<grid1((long long)kEE * kED), 256, 0, stream>>>(z_b, e_f, e_b);
    agg_p_kernel<<<grid1(kNN * 3), 256, 0, stream>>>(rn, gp, p_f);
  }

  // ---- head
  gemm_bf16_kernel<<<dim3(kNN / 128, kSD / 32), 256, 0, stream>>>(
      s_b, wshb, b_sh, sh_f, nullptr, kNN, kSD, kSD, 1);               // sh = silu(...)
  coords_kernel<<<grid1(kNN * 3), 256, 0, stream>>>(p_f, v_f, W_cl, coords);
  center_kernel<<<kNB, kMOL, 0, stream>>>(coords, coords);
  esym_kernel<<<grid1((long long)kEE * kED), 256, 0, stream>>>(e_f, esym_b);
  gemm_bf16_kernel<<<dim3(kEE / 128, kSD / 32), 256, 0, stream>>>(
      esym_b, wbmapb, b_bmap, nullptr, f0_b, kEE, kSD, kED, 0);        // f0 = esym@Wbmap + b
  combine_kernel<<<grid1((long long)kEE * kKB0), 256, 0, stream>>>(
      f0_b, sh_f, srcI, dstI, coords, ein_b);
  gemm_bf16_kernel<<<dim3(kEE / 128, kSD / 32), 256, 0, stream>>>(
      ein_b, wb0b, b_b0, nullptr, g_b, kEE, kSD, kKB0, 1);             // silu(edge@W_b0+b)
  gemm_bf16_kernel<<<dim3(kEE / 128, kNB1 / 32), 256, 0, stream>>>(
      g_b, wb1b, bb1p, out32, nullptr, kEE, kNB1, kSD, 0);             // @W_b1 + b_b1
  out_copy_kernel<<<grid1((long long)kEE * 5), 256, 0, stream>>>(out32, out);
}
